// VersorTransformer_62113817034791
// MI455X (gfx1250) — matrix operand order repaired and checked by reading, not compiled, not run on hardware
//
#include <hip/hip_runtime.h>
#include <cstddef>

// ---------------------------------------------------------------------------
// Versor Transformer forward for MI455X (gfx1250, wave32, WMMA).
// GEMMs on v_wmma_f32_16x16x32_bf16 (f32 accum, dual accumulators to break
// WMMA RAW chains); A fragments are contiguous b128 loads, B fragments u16 gathers. The one
// transposed operand (K in the scores GEMM) is pre-transposed once per layer
// through an LDS-tiled transpose. Per-multivector (G=32) ops map
// 1 multivector -> 1 wave32 with shfl reductions.
// Workspace requirement: 129 MB.
// ---------------------------------------------------------------------------

typedef __attribute__((ext_vector_type(16))) __bf16        v16bf;
typedef __attribute__((ext_vector_type(8)))  float         v8f;
typedef __attribute__((ext_vector_type(4)))  unsigned int  u32x4;
typedef __attribute__((ext_vector_type(2)))  unsigned int  u32x2;

union BF16x16 { v16bf v; u32x4 q[2]; unsigned short u[16]; };
union BF16x8  { u32x4 q; unsigned short u[8]; };

__device__ __forceinline__ unsigned short f2bf(float f) {
  unsigned int u = __float_as_uint(f);
  u = (u + 0x7FFFu + ((u >> 16) & 1u)) >> 16;           // round-to-nearest-even
  return (unsigned short)u;
}

__device__ __forceinline__ v8f wmma_bf16(v16bf a, v16bf b, v8f c) {
  // (neg_a, A, neg_b, B, c_mod, C, reuse_a, reuse_b)
  return __builtin_amdgcn_wmma_f32_16x16x32_bf16(false, a, false, b, (short)0, c,
                                                 false, false);
}

__device__ __forceinline__ float wave_sum(float v) {
  #pragma unroll
  for (int o = 16; o; o >>= 1) v += __shfl_xor(v, o, 32);
  return v;
}

// ---------------------------------------------------------------------------
// f32 -> bf16 converter, 4 elements/thread (n must be a multiple of 4).
// ---------------------------------------------------------------------------
__global__ __launch_bounds__(256) void f32_to_bf16x4(
    const float* __restrict__ in, unsigned short* __restrict__ out, int n4) {
  const int i = blockIdx.x * blockDim.x + threadIdx.x;
  if (i < n4) {
    const float4 v = ((const float4*)in)[i];
    u32x2 r;
    r.x = (unsigned int)f2bf(v.x) | ((unsigned int)f2bf(v.y) << 16);
    r.y = (unsigned int)f2bf(v.z) | ((unsigned int)f2bf(v.w) << 16);
    ((u32x2*)out)[i] = r;
  }
}

// ---------------------------------------------------------------------------
// LayerNorm2d over (D,G)=2048 per token, emit bf16.  grid = 2048 tokens,
// 256 threads, 8 contiguous elements per thread (b128 in, b128 out).
// ---------------------------------------------------------------------------
__global__ __launch_bounds__(256) void ln2d_bf16_kernel(
    const float* __restrict__ x, const float* __restrict__ w,
    const float* __restrict__ b, unsigned short* __restrict__ out) {
  const int t = blockIdx.x;
  const int base = threadIdx.x * 8;
  const float4* xt = (const float4*)(x + (size_t)t * 2048 + base);
  const float4 v0 = xt[0], v1 = xt[1];
  float vals[8] = {v0.x, v0.y, v0.z, v0.w, v1.x, v1.y, v1.z, v1.w};
  float s = 0.f, ss = 0.f;
  #pragma unroll
  for (int i = 0; i < 8; ++i) { s += vals[i]; ss += vals[i] * vals[i]; }
  s = wave_sum(s); ss = wave_sum(ss);
  __shared__ float red0[8], red1[8];
  __shared__ float smu, srstd;
  const int wid = threadIdx.x >> 5, lane = threadIdx.x & 31;
  if (lane == 0) { red0[wid] = s; red1[wid] = ss; }
  __syncthreads();
  if (threadIdx.x == 0) {
    float S0 = 0.f, S1 = 0.f;
    #pragma unroll
    for (int i = 0; i < 8; ++i) { S0 += red0[i]; S1 += red1[i]; }
    const float mu = S0 * (1.f / 2048.f);
    smu = mu; srstd = rsqrtf(S1 * (1.f / 2048.f) - mu * mu + 1e-5f);
  }
  __syncthreads();
  const float mu = smu, rstd = srstd;
  const float4* wp = (const float4*)(w + base);
  const float4* bp = (const float4*)(b + base);
  const float4 w0 = wp[0], w1 = wp[1], b0 = bp[0], b1 = bp[1];
  const float wv[8] = {w0.x, w0.y, w0.z, w0.w, w1.x, w1.y, w1.z, w1.w};
  const float bv[8] = {b0.x, b0.y, b0.z, b0.w, b1.x, b1.y, b1.z, b1.w};
  BF16x8 o;
  #pragma unroll
  for (int i = 0; i < 8; ++i) o.u[i] = f2bf((vals[i] - mu) * rstd * wv[i] + bv[i]);
  *(u32x4*)(out + (size_t)t * 2048 + base) = o.q;
}

// ---------------------------------------------------------------------------
// Token linear: Y[t,o,g] = sum_i W[o,i]*X[t,i,g] + bias[o,g]
// One wave -> 16(O) x 32(G) tile of one token, dual K-parity accumulators.
// MODE 0: f32 out, 1: bf16 out, 2: tanh->bf16 out. Din % 64 == 0.
// ---------------------------------------------------------------------------
template <int MODE>
__global__ __launch_bounds__(256) void tok_linear_wmma(
    const unsigned short* __restrict__ Wbf, const float* __restrict__ bias,
    const unsigned short* __restrict__ Xbf,
    float* __restrict__ Yf, unsigned short* __restrict__ Ybf,
    int Din, int O) {
  const int wave = (blockIdx.x * blockDim.x + threadIdx.x) >> 5;
  const int lane = threadIdx.x & 31;
  const int tilesPerTok = O >> 4;
  const int t  = wave / tilesPerTok;
  const int m0 = (wave % tilesPerTok) << 4;
  const int row   = lane & 15;
  const int base0 = (lane >> 4) << 3;                  // A half-wave K offset
  const int kb    = (lane & 15) + ((lane >> 4) << 4);  // not used
  const size_t FX = (size_t)Din * 32;
  v8f acc0a = {}, acc0b = {}, acc1a = {}, acc1b = {};
  for (int k0 = 0; k0 < Din; k0 += 64) {
    BF16x16 aA, aB, b0, b1, b2, b3;
    const unsigned short* wr = Wbf + (size_t)(m0 + row) * Din + k0;
    aA.q[0] = *(const u32x4*)(wr + base0);
    aA.q[1] = *(const u32x4*)(wr + 16 + base0);
    aB.q[0] = *(const u32x4*)(wr + 32 + base0);
    aB.q[1] = *(const u32x4*)(wr + 48 + base0);
    const unsigned short* xt = Xbf + (size_t)t * FX + (size_t)k0 * 32 + row;
    #pragma unroll
    for (int s = 0; s < 16; ++s) {
      const int kk = base0 + (s & 7) + ((s >> 3) << 4);
      b0.u[s] = xt[(size_t)kk * 32];
      b1.u[s] = xt[(size_t)kk * 32 + 16];
      b2.u[s] = xt[(size_t)(32 + kk) * 32];
      b3.u[s] = xt[(size_t)(32 + kk) * 32 + 16];
    }
    acc0a = wmma_bf16(aA.v, b0.v, acc0a);
    acc1a = wmma_bf16(aA.v, b1.v, acc1a);
    acc0b = wmma_bf16(aB.v, b2.v, acc0b);
    acc1b = wmma_bf16(aB.v, b3.v, acc1b);
  }
  const int n = lane & 15;
  const int mh = (lane >> 4) << 3;
  #pragma unroll
  for (int r = 0; r < 8; ++r) {
    const int m = m0 + mh + r;
    float v0 = acc0a[r] + acc0b[r] + bias[m * 32 + n];
    float v1 = acc1a[r] + acc1b[r] + bias[m * 32 + 16 + n];
    const size_t o0 = (size_t)t * O * 32 + (size_t)m * 32 + n;
    if (MODE == 0)      { Yf[o0] = v0;               Yf[o0 + 16] = v1; }
    else if (MODE == 1) { Ybf[o0] = f2bf(v0);        Ybf[o0 + 16] = f2bf(v1); }
    else                { Ybf[o0] = f2bf(tanhf(v0)); Ybf[o0 + 16] = f2bf(tanhf(v1)); }
  }
}

// ---------------------------------------------------------------------------
// K transpose: KT[bh][f][t] = K[b,t][h*256+f]. LDS 16x16 tile per wave:
// coalesced b128 global loads/stores; scatter confined to 8 ds_load_u16/lane.
// One wave per 16x16 tile; grid = 64*16*16 = 16384 waves.
// ---------------------------------------------------------------------------
__global__ __launch_bounds__(256) void ktranspose_kernel(
    const unsigned short* __restrict__ Kbf, unsigned short* __restrict__ KT) {
  __shared__ unsigned short tile[8][16][16];
  const int wave = (blockIdx.x * blockDim.x + threadIdx.x) >> 5;
  const int lane = threadIdx.x & 31;
  const int wi = threadIdx.x >> 5;
  const int tt = wave & 15, tf = (wave >> 4) & 15, bh = wave >> 8;
  const int b = bh >> 3, h = bh & 7;
  const size_t tokbase = (size_t)b * 256 * 2048 + (size_t)h * 256;
  const int t0 = tt << 4, f0 = tf << 4;
  const int lt = lane & 15, half = (lane >> 4) << 3;
  // coalesced: token t0+lt, 8 contiguous features f0+half..
  *(u32x4*)&tile[wi][lt][half] =
      *(const u32x4*)(Kbf + tokbase + (size_t)(t0 + lt) * 2048 + f0 + half);
  __builtin_amdgcn_wave_barrier();
  BF16x8 o;
  #pragma unroll
  for (int i = 0; i < 8; ++i) o.u[i] = tile[wi][half + i][lt];
  __builtin_amdgcn_wave_barrier();
  *(u32x4*)(KT + (size_t)bh * 65536 + (size_t)(f0 + lt) * 256 + t0 + half) = o.q;
}

// ---------------------------------------------------------------------------
// Attention scores: per (b,h): S[s,t] = scale * sum_f Q[s,f]*KT[f,t].
// A operands contiguous b128, B gathered per u16 element; dual K-parity accumulators.
// ---------------------------------------------------------------------------
__global__ __launch_bounds__(256) void attn_scores_wmma(
    const unsigned short* __restrict__ Qbf, const unsigned short* __restrict__ KT,
    float* __restrict__ scores, float scale) {
  const int wave = (blockIdx.x * blockDim.x + threadIdx.x) >> 5;
  const int lane = threadIdx.x & 31;
  const int nt = wave & 15, mt = (wave >> 4) & 15, bh = wave >> 8;
  const int b = bh >> 3, h = bh & 7;
  const size_t tokbase = (size_t)b * 256 * 2048 + (size_t)h * 256;
  const int s0 = mt << 4, t0 = nt << 4;
  const int row   = lane & 15;
  const int base0 = (lane >> 4) << 3;
  const int kb    = (lane & 15) + ((lane >> 4) << 4);
  const unsigned short* qr = Qbf + tokbase + (size_t)(s0 + row) * 2048;
  const unsigned short* kc = KT + (size_t)bh * 65536 + t0;
  v8f accA = {}, accB = {};
  for (int f0 = 0; f0 < 256; f0 += 64) {
    BF16x16 aA, aB, bA, bB;
    aA.q[0] = *(const u32x4*)(qr + f0 + base0);
    aA.q[1] = *(const u32x4*)(qr + f0 + 16 + base0);
    aB.q[0] = *(const u32x4*)(qr + f0 + 32 + base0);
    aB.q[1] = *(const u32x4*)(qr + f0 + 48 + base0);
    #pragma unroll
    for (int s = 0; s < 16; ++s) {
      const int kk = base0 + (s & 7) + ((s >> 3) << 4);
      bA.u[s] = kc[(size_t)(f0 + kk) * 256 + row];
      bB.u[s] = kc[(size_t)(f0 + 32 + kk) * 256 + row];
    }
    accA = wmma_bf16(aA.v, bA.v, accA);
    accB = wmma_bf16(aB.v, bB.v, accB);
  }
  const int n = lane & 15, mh = (lane >> 4) << 3;
  float* srow = scores + (size_t)bh * 65536;
  #pragma unroll
  for (int r = 0; r < 8; ++r)
    srow[(size_t)(s0 + mh + r) * 256 + t0 + n] = (accA[r] + accB[r]) * scale;
}

// ---------------------------------------------------------------------------
// Row softmax over 256 entries; one wave per row; emit bf16 probs.
// ---------------------------------------------------------------------------
__global__ __launch_bounds__(256) void softmax_rows(
    const float* __restrict__ scores, unsigned short* __restrict__ probs) {
  const int wave = (blockIdx.x * blockDim.x + threadIdx.x) >> 5;
  const int lane = threadIdx.x & 31;
  const float4* rowp = (const float4*)(scores + (size_t)wave * 256);
  float4 c0 = rowp[lane], c1 = rowp[lane + 32];
  float v[8] = {c0.x, c0.y, c0.z, c0.w, c1.x, c1.y, c1.z, c1.w};
  float mx = -1e30f;
  #pragma unroll
  for (int i = 0; i < 8; ++i) mx = fmaxf(mx, v[i]);
  #pragma unroll
  for (int o = 16; o; o >>= 1) mx = fmaxf(mx, __shfl_xor(mx, o, 32));
  float sum = 0.f;
  #pragma unroll
  for (int i = 0; i < 8; ++i) { v[i] = __expf(v[i] - mx); sum += v[i]; }
  sum = wave_sum(sum);
  const float inv = 1.f / sum;
  BF16x8 o0, o1;
  #pragma unroll
  for (int i = 0; i < 4; ++i) { o0.u[i] = f2bf(v[i] * inv); o1.u[i] = f2bf(v[4 + i] * inv); }
  unsigned short* op = probs + (size_t)wave * 256;
  *(u32x2*)(op + lane * 4)       = u32x2{o0.q.x, o0.q.y};
  *(u32x2*)(op + 128 + lane * 4) = u32x2{o1.q.x, o1.q.y};
}

// ---------------------------------------------------------------------------
// PV: per (b,h): O[s,f] = sum_t P[s,t]*V[t,f]; emit bf16 in token layout.
// A operands contiguous b128, B gathered per u16 element; dual K-parity accumulators.
// ---------------------------------------------------------------------------
__global__ __launch_bounds__(256) void attn_pv_wmma(
    const unsigned short* __restrict__ probs, const unsigned short* __restrict__ Vbf,
    unsigned short* __restrict__ Obf) {
  const int wave = (blockIdx.x * blockDim.x + threadIdx.x) >> 5;
  const int lane = threadIdx.x & 31;
  const int nt = wave & 15, mt = (wave >> 4) & 15, bh = wave >> 8;
  const int b = bh >> 3, h = bh & 7;
  const size_t tokbase = (size_t)b * 256 * 2048 + (size_t)h * 256;
  const unsigned short* P = probs + (size_t)bh * 65536;
  const int s0 = mt << 4, f0 = nt << 4;
  const int row   = lane & 15;
  const int base0 = (lane >> 4) << 3;
  const int kb    = (lane & 15) + ((lane >> 4) << 4);
  v8f accA = {}, accB = {};
  for (int k0 = 0; k0 < 256; k0 += 64) {
    BF16x16 aA, aB, bA, bB;
    const unsigned short* pr = P + (size_t)(s0 + row) * 256 + k0;
    aA.q[0] = *(const u32x4*)(pr + base0);
    aA.q[1] = *(const u32x4*)(pr + 16 + base0);
    aB.q[0] = *(const u32x4*)(pr + 32 + base0);
    aB.q[1] = *(const u32x4*)(pr + 48 + base0);
    #pragma unroll
    for (int s = 0; s < 16; ++s) {
      const int kk = base0 + (s & 7) + ((s >> 3) << 4);
      bA.u[s] = Vbf[tokbase + (size_t)(k0 + kk) * 2048 + f0 + row];
      bB.u[s] = Vbf[tokbase + (size_t)(k0 + 32 + kk) * 2048 + f0 + row];
    }
    accA = wmma_bf16(aA.v, bA.v, accA);
    accB = wmma_bf16(aB.v, bB.v, accB);
  }
  const int n = lane & 15, mh = (lane >> 4) << 3;
  #pragma unroll
  for (int r = 0; r < 8; ++r)
    Obf[tokbase + (size_t)(s0 + mh + r) * 2048 + f0 + n] = f2bf(accA[r] + accB[r]);
}

// ---------------------------------------------------------------------------
// Residual + manifold_norm over G=32: one wave per (token, d). In-place on x.
// ---------------------------------------------------------------------------
__global__ __launch_bounds__(256) void resnorm_kernel(
    float* __restrict__ x, const float* __restrict__ h,
    const float* __restrict__ gamma) {
  const int wave = (blockIdx.x * blockDim.x + threadIdx.x) >> 5;
  const int lane = threadIdx.x & 31;
  const size_t idx = (size_t)wave * 32 + lane;
  const int d = wave & 63;
  const float v = x[idx] + gamma[d * 32 + lane] * h[idx];
  const float ss = wave_sum(v * v);
  x[idx] = v / (sqrtf(ss) + 1e-6f);
}

// rot = manifold_norm(0.5*delta + e0); one wave per multivector.
__global__ __launch_bounds__(256) void make_rot_kernel(
    const float* __restrict__ delta, float* __restrict__ rot) {
  const int wave = (blockIdx.x * blockDim.x + threadIdx.x) >> 5;
  const int lane = threadIdx.x & 31;
  const size_t idx = (size_t)wave * 32 + lane;
  const float v = 0.5f * delta[idx] + (lane == 0 ? 1.f : 0.f);
  const float ss = wave_sum(v * v);
  rot[idx] = v / (sqrtf(ss) + 1e-6f);
}

// Cayley sign for Cl(4,1): SIG = [1,1,1,1,-1]; blades < 32.
__device__ __forceinline__ float gp_sign(int a, int b) {
  int s = __popc((a >> 1) & b) + __popc((a >> 2) & b) +
          __popc((a >> 3) & b) + __popc((a >> 4) & b);
  s += (a & b) >> 4;   // e4 has signature -1
  return (s & 1) ? -1.f : 1.f;
}

// One doubling step of the GP scan: out[s] = mn(gp(in[s], in[s-step])) for s>=step.
__global__ __launch_bounds__(256) void scan_step_kernel(
    const float* __restrict__ rin, float* __restrict__ rout, int step) {
  const int wave = (blockIdx.x * blockDim.x + threadIdx.x) >> 5;  // (b*256+s)*64+d
  const int lane = threadIdx.x & 31;                              // lane = blade k
  const size_t idx = (size_t)wave * 32 + lane;
  const int s = (wave >> 6) & 255;
  const float rv = rin[idx];
  if (s < step) { rout[idx] = rv; return; }
  const float lv = rin[idx - (size_t)step * 2048];
  const int k = lane;
  float ok = 0.f;
  for (int a = 0; a < 32; ++a) {
    const float ra = __shfl(rv, a, 32);
    const int bi = a ^ k;
    const float lb = __shfl(lv, bi, 32);
    ok += ra * lb * gp_sign(a, bi);
  }
  const float ss = wave_sum(ok * ok);
  rout[idx] = ok / (sqrtf(ss) + 1e-6f);
}

// logits[b,c] = dot(rot[b, S-1, :], Wc[c,:]) + bc[c]; one wave per output.
__global__ __launch_bounds__(256) void classifier_kernel(
    const float* __restrict__ rot, const float* __restrict__ Wc,
    const float* __restrict__ bc, float* __restrict__ out) {
  const int wave = (blockIdx.x * blockDim.x + threadIdx.x) >> 5;  // b*1000+c
  const int lane = threadIdx.x & 31;
  const int b = wave / 1000, c = wave % 1000;
  const float4* p4 = (const float4*)(rot + ((size_t)b * 256 + 255) * 2048);
  const float4* w4 = (const float4*)(Wc + (size_t)c * 2048);
  float acc = 0.f;
  #pragma unroll 4
  for (int i = lane; i < 512; i += 32) {
    const float4 a = p4[i], w = w4[i];
    acc += a.x * w.x + a.y * w.y + a.z * w.z + a.w * w.w;
  }
  acc = wave_sum(acc);
  if (lane == 0) out[wave] = acc + bc[c];
}

// ---------------------------------------------------------------------------
// Host orchestration
// ---------------------------------------------------------------------------
extern "C" void kernel_launch(void* const* d_in, const int* in_sizes, int n_in,
                              void* d_out, int out_size, void* d_ws, size_t ws_size,
                              hipStream_t stream) {
  (void)in_sizes; (void)n_in; (void)out_size; (void)ws_size;
  const float* x_in = (const float*)d_in[0];
  const float* Wq = (const float*)d_in[1];  const float* bq = (const float*)d_in[2];
  const float* Wk = (const float*)d_in[3];  const float* bk = (const float*)d_in[4];
  const float* Wv = (const float*)d_in[5];  const float* bv = (const float*)d_in[6];
  const float* Wo = (const float*)d_in[7];  const float* bo = (const float*)d_in[8];
  const float* ln1w = (const float*)d_in[9];  const float* ln1b = (const float*)d_in[10];
  const float* ln2w = (const float*)d_in[11]; const float* ln2b = (const float*)d_in[12];
  const float* g1 = (const float*)d_in[13]; const float* g2 = (const float*)d_in[14];
  const float* W1 = (const float*)d_in[15]; const float* b1 = (const float*)d_in[16];
  const float* W2 = (const float*)d_in[17]; const float* b2 = (const float*)d_in[18];
  const float* Wr = (const float*)d_in[19]; const float* br = (const float*)d_in[20];
  const float* Wc = (const float*)d_in[21]; const float* bc = (const float*)d_in[22];
  float* out = (float*)d_out;

  char* ws = (char*)d_ws;
  const size_t MB = 1u << 20;
  float*          X    = (float*)(ws + 0);                 // 16 MB  (T=2048, F=2048) f32
  unsigned short* XNBF = (unsigned short*)(ws + 16 * MB);  //  8 MB  bf16 normalized x / x
  unsigned short* QBF  = (unsigned short*)(ws + 24 * MB);  //  8 MB
  unsigned short* KBF  = (unsigned short*)(ws + 32 * MB);  //  8 MB
  unsigned short* VBF  = (unsigned short*)(ws + 40 * MB);  //  8 MB
  float*          SC   = (float*)(ws + 48 * MB);           // 16 MB  scores (later ROT_A)
  unsigned short* PBF  = (unsigned short*)(ws + 64 * MB);  //  8 MB  probs
  unsigned short* OBF  = (unsigned short*)(ws + 72 * MB);  //  8 MB  attn out
  float*          HF   = (float*)(ws + 80 * MB);           // 16 MB  h / m2 / delta
  unsigned short* MBF  = (unsigned short*)(ws + 96 * MB);  // 32 MB  mlp hidden
  unsigned short* KT   = (unsigned short*)(ws + 96 * MB);  //  8 MB  K^T (dead when MBF live)
  float*          ROTA = SC;
  float*          ROTB = (float*)(ws + 96 * MB);
  // bf16 weight cache (392 KB) at +128 MB
  unsigned short* WQBF = (unsigned short*)(ws + 128 * MB);
  unsigned short* WKBF = WQBF + 16384;
  unsigned short* WVBF = WKBF + 16384;
  unsigned short* WOBF = WVBF + 16384;
  unsigned short* W1BF = WOBF + 16384;
  unsigned short* W2BF = W1BF + 65536;
  unsigned short* WRBF = W2BF + 65536;

  hipMemcpyAsync(X, x_in, (size_t)2048 * 2048 * 4, hipMemcpyDeviceToDevice, stream);

  // one-time bf16 weight conversion (all layers)
  f32_to_bf16x4<<<16, 256, 0, stream>>>(Wq, WQBF, 4096);
  f32_to_bf16x4<<<16, 256, 0, stream>>>(Wk, WKBF, 4096);
  f32_to_bf16x4<<<16, 256, 0, stream>>>(Wv, WVBF, 4096);
  f32_to_bf16x4<<<16, 256, 0, stream>>>(Wo, WOBF, 4096);
  f32_to_bf16x4<<<64, 256, 0, stream>>>(W1, W1BF, 16384);
  f32_to_bf16x4<<<64, 256, 0, stream>>>(W2, W2BF, 16384);
  f32_to_bf16x4<<<4, 256, 0, stream>>>(Wr, WRBF, 1024);

  const float scale = 0.0625f;  // 1/sqrt(dh*G) = 1/16
  for (int l = 0; l < 4; ++l) {
    const size_t wOff = (size_t)l * 64 * 64, bOff = (size_t)l * 2048;
    // --- attention block ---
    ln2d_bf16_kernel<<<2048, 256, 0, stream>>>(X, ln1w + bOff, ln1b + bOff, XNBF);
    tok_linear_wmma<1><<<1024, 256, 0, stream>>>(WQBF + wOff, bq + bOff, XNBF, nullptr, QBF, 64, 64);
    tok_linear_wmma<1><<<1024, 256, 0, stream>>>(WKBF + wOff, bk + bOff, XNBF, nullptr, KBF, 64, 64);
    tok_linear_wmma<1><<<1024, 256, 0, stream>>>(WVBF + wOff, bv + bOff, XNBF, nullptr, VBF, 64, 64);
    ktranspose_kernel<<<2048, 256, 0, stream>>>(KBF, KT);
    attn_scores_wmma<<<2048, 256, 0, stream>>>(QBF, KT, SC, scale);
    softmax_rows<<<2048, 256, 0, stream>>>(SC, PBF);
    attn_pv_wmma<<<2048, 256, 0, stream>>>(PBF, VBF, OBF);
    tok_linear_wmma<0><<<1024, 256, 0, stream>>>(WOBF + wOff, bo + bOff, OBF, HF, nullptr, 64, 64);
    resnorm_kernel<<<16384, 256, 0, stream>>>(X, HF, g1 + bOff);
    // --- MLP block ---
    ln2d_bf16_kernel<<<2048, 256, 0, stream>>>(X, ln2w + bOff, ln2b + bOff, XNBF);
    tok_linear_wmma<2><<<4096, 256, 0, stream>>>(W1BF + (size_t)l * 256 * 64, b1 + (size_t)l * 8192,
                                                 XNBF, nullptr, MBF, 64, 256);
    tok_linear_wmma<0><<<1024, 256, 0, stream>>>(W2BF + (size_t)l * 64 * 256, b2 + bOff,
                                                 MBF, HF, nullptr, 256, 64);
    resnorm_kernel<<<16384, 256, 0, stream>>>(X, HF, g2 + bOff);
  }

  // --- rotor head + GP scan + classifier ---
  f32_to_bf16x4<<<4096, 256, 0, stream>>>(X, XNBF, 1048576);
  tok_linear_wmma<0><<<1024, 256, 0, stream>>>(WRBF, br, XNBF, HF, nullptr, 64, 64);
  make_rot_kernel<<<16384, 256, 0, stream>>>(HF, ROTA);
  float* pin = ROTA; float* pout = ROTB;
  for (int step = 1; step < 256; step <<= 1) {
    scan_step_kernel<<<16384, 256, 0, stream>>>(pin, pout, step);
    float* tmp = pin; pin = pout; pout = tmp;
  }
  classifier_kernel<<<1000, 256, 0, stream>>>(pin, Wc, bc, out);
}
